// FullVAE_Hist_GLM_71683004170394
// MI455X (gfx1250) — compile-verified, matrix operand order checked by reading, not run on hardware
//
#include <hip/hip_runtime.h>
#include <hip/hip_bf16.h>

// ---------------------------------------------------------------------------
// Problem constants (from reference)
// ---------------------------------------------------------------------------
#define T_DATA_  100000
#define SUB_NO_  20
#define T_NO_    200
#define NB_      19
#define E_NO_    2000
#define I_NO_    400
#define SPAD_    32      // padded subunit stride for intermediates

typedef __bf16 bf16_t;
typedef __attribute__((ext_vector_type(16))) __bf16    v16bf;
typedef __attribute__((ext_vector_type(8)))  float     v8f;
typedef __attribute__((ext_vector_type(4)))  float     f32x4;
typedef __attribute__((ext_vector_type(4)))  unsigned  u32x4;

// Pack two f32 into one dword of two bf16 (truncation — exact for 0/1 data).
// v_perm_b32: out bytes {hi.b3, hi.b2, lo.b3, lo.b2}
__device__ __forceinline__ unsigned pack2bf(float lo, float hi) {
  return __builtin_amdgcn_perm(__float_as_uint(hi), __float_as_uint(lo),
                               0x07060302u);
}

union Frag16 {            // one 16-element bf16 WMMA fragment = 8 dwords
  unsigned u[8];
  v16bf    v;
};

// ---------------------------------------------------------------------------
// Kernel 1: build e/i/hist/spk filter banks [4*20, 200]
// ---------------------------------------------------------------------------
__global__ void __launch_bounds__(256) filters_kernel(
    const float* __restrict__ Wsyn, const float* __restrict__ Whist,
    const float* __restrict__ Wspk, const float* __restrict__ Delta,
    const float* __restrict__ Tau,  float* __restrict__ out_filters,
    float* __restrict__ ws_kern) {
  int idx = blockIdx.x * 256 + threadIdx.x;
  if (idx >= 4 * SUB_NO_ * T_NO_) return;
  int x   = idx % T_NO_;
  int row = idx / T_NO_;
  int grp = row / SUB_NO_;
  int s   = row % SUB_NO_;
  float val;
  if (grp == 3) {
    float tt   = fmaxf((float)x - Delta[s], 0.0f);
    float tau  = Tau[s];
    float ttau = tt / (tau * tau);
    float w    = Wspk[s];
    val = ttau * __expf(-ttau) * (w * w);
  } else {
    const float PI = 3.14159265358979f;
    float raw = 5.0f * __logf((float)x + 1.0f);
    float acc = 0.0f;
#pragma unroll
    for (int b = 0; b < NB_; ++b) {
      float phi = 0.5f * PI * (float)b;
      float d   = raw - phi;
      float cb  = (d >= -PI && d <= PI) ? (0.5f * __cosf(d) + 0.5f) : 0.0f;
      float w   = (grp == 0) ? Wsyn[s * (NB_ * 2) + b * 2 + 0]
                : (grp == 1) ? Wsyn[s * (NB_ * 2) + b * 2 + 1]
                             : Whist[s * NB_ + b];
      acc += w * cb;
    }
    val = acc;
  }
  out_filters[idx] = val;
  ws_kern[idx]     = val;
}

// ---------------------------------------------------------------------------
// Kernel 2a: pre-pack B fragments (C_syn^T tiles) into WMMA bf16 layout.
// Layout: Bfrag[((ntile*nchunks + kc)*32 + lane)*8 + d]  (dwords)
// Element e of lane's v16bf:  n = ntile*16 + (lane&15),
//   k = kc*32 + ((lane>>4)*8) + (e<8 ? e : 16 + (e-8))
// Out-of-range n/k packed as 0 here so the hot loop needs no guards.
// ---------------------------------------------------------------------------
__global__ void __launch_bounds__(256) pack_b_kernel(
    const float* __restrict__ Cmap, unsigned* __restrict__ Bfrag,
    int Ktot, int nchunks) {
  int idx = blockIdx.x * 256 + threadIdx.x;
  int total = 2 * nchunks * 32 * 8;
  if (idx >= total) return;
  int d     = idx & 7;
  int lane  = (idx >> 3) & 31;
  int kc    = (idx >> 8) % nchunks;
  int ntile = (idx >> 8) / nchunks;
  int koff  = (lane >> 4) << 3;
  int n     = ntile * 16 + (lane & 15);
  int e0    = 2 * d;
  int k0    = kc * 32 + koff + (e0 < 8 ? e0 : 16 + (e0 - 8));
  int k1    = k0 + 1;
  float f0 = (n < SUB_NO_ && k0 < Ktot) ? Cmap[(size_t)n * Ktot + k0] : 0.0f;
  float f1 = (n < SUB_NO_ && k1 < Ktot) ? Cmap[(size_t)n * Ktot + k1] : 0.0f;
  Bfrag[idx] = pack2bf(f0, f1);
}

// ---------------------------------------------------------------------------
// Kernel 2b: syn = S[T,K] @ Cmap[N,K]^T via bf16 WMMA.
// One wave handles a 32(time) x 16(subunit) strip: 2 WMMAs per 32-K chunk
// sharing one pre-packed B fragment. A is streamed with non-temporal b128
// loads and packed to bf16 with v_perm (1 VALU per dword). K%32==16 tail:
// group0 vector-loaded (in-bounds), group1 zero (B tail already zero).
// ---------------------------------------------------------------------------
__device__ __forceinline__ void load_a_group_nt(const float* p, unsigned* d) {
  f32x4 x0 = __builtin_nontemporal_load((const f32x4*)p);
  f32x4 x1 = __builtin_nontemporal_load((const f32x4*)(p + 4));
  d[0] = pack2bf(x0.x, x0.y);
  d[1] = pack2bf(x0.z, x0.w);
  d[2] = pack2bf(x1.x, x1.y);
  d[3] = pack2bf(x1.z, x1.w);
}

__global__ void __launch_bounds__(32) agg_wmma_kernel(
    const float* __restrict__ S, const unsigned* __restrict__ Bfrag,
    float* __restrict__ dst /* [T, SPAD_] */, int Ktot, int nchunks) {
  const int lane  = threadIdx.x;
  const int m     = lane & 15;
  const int koff  = (lane >> 4) << 3;          // 0 or 8 (ISA 16-bit A layout)
  const int ntile = blockIdx.x & 1;            // subunit tile (0..1, padded)
  const int t0    = (blockIdx.x >> 1) << 5;    // 32 time rows per wave

  const float* __restrict__ ar0 = S + (size_t)(t0 + m) * (size_t)Ktot;
  const float* __restrict__ ar1 = S + (size_t)(t0 + 16 + m) * (size_t)Ktot;
  const unsigned* __restrict__ bp =
      Bfrag + ((size_t)ntile * nchunks * 32 + (size_t)lane) * 8;

  const int nfull = Ktot >> 5;
  v8f c0 = {}, c1 = {};

  for (int kc = 0; kc < nfull; ++kc) {
    const int kb = (kc << 5) + koff;
    // B fragment: 2x b128 from pre-packed, L2-resident table
    Frag16 b;
    const u32x4* bq = (const u32x4*)(bp + (size_t)kc * 256);
    u32x4 bl = bq[0], bh = bq[1];
    b.u[0] = bl.x; b.u[1] = bl.y; b.u[2] = bl.z; b.u[3] = bl.w;
    b.u[4] = bh.x; b.u[5] = bh.y; b.u[6] = bh.z; b.u[7] = bh.w;
    // A fragments: 4x non-temporal b128 each strip-row, packed via v_perm
    Frag16 a0, a1;
    load_a_group_nt(ar0 + kb,      a0.u);
    load_a_group_nt(ar0 + kb + 16, a0.u + 4);
    load_a_group_nt(ar1 + kb,      a1.u);
    load_a_group_nt(ar1 + kb + 16, a1.u + 4);
    __builtin_prefetch(ar0 + kb + 256, 0, 3);
    c0 = __builtin_amdgcn_wmma_f32_16x16x32_bf16(false, a0.v, false, b.v,
                                                 (short)0, c0, false, false);
    c1 = __builtin_amdgcn_wmma_f32_16x16x32_bf16(false, a1.v, false, b.v,
                                                 (short)0, c1, false, false);
  }

  if (Ktot & 31) {  // tail chunk (remainder == 16 for E_NO/I_NO)
    const int kb = (nfull << 5) + koff;
    Frag16 b;
    const u32x4* bq = (const u32x4*)(bp + (size_t)nfull * 256);
    u32x4 bl = bq[0], bh = bq[1];
    b.u[0] = bl.x; b.u[1] = bl.y; b.u[2] = bl.z; b.u[3] = bl.w;
    b.u[4] = bh.x; b.u[5] = bh.y; b.u[6] = bh.z; b.u[7] = bh.w;
    Frag16 a0, a1;
    load_a_group_nt(ar0 + kb, a0.u);
    load_a_group_nt(ar1 + kb, a1.u);
#pragma unroll
    for (int j = 4; j < 8; ++j) { a0.u[j] = 0u; a1.u[j] = 0u; }
    c0 = __builtin_amdgcn_wmma_f32_16x16x32_bf16(false, a0.v, false, b.v,
                                                 (short)0, c0, false, false);
    c1 = __builtin_amdgcn_wmma_f32_16x16x32_bf16(false, a1.v, false, b.v,
                                                 (short)0, c1, false, false);
  }

  // C/D layout: VGPR r -> M = r + (lane>=16 ? 8 : 0), N = lane&15
  float* __restrict__ out = dst + (size_t)t0 * SPAD_ + (size_t)(ntile * 16 + m);
  const int mrow = (lane < 16) ? 0 : 8;
#pragma unroll
  for (int r = 0; r < 8; ++r)
    out[(size_t)(r + mrow) * SPAD_] = c0[r];
  out += (size_t)16 * SPAD_;
#pragma unroll
  for (int r = 0; r < 8; ++r)
    out[(size_t)(r + mrow) * SPAD_] = c1[r];
}

// ---------------------------------------------------------------------------
// Kernel 3: fused depthwise causal convs (e, i, hist, spk) per channel.
// hist/spk use Z delayed by one step (reference pads T_no and drops last).
// ---------------------------------------------------------------------------
__global__ void __launch_bounds__(256) conv_kernel(
    const float* __restrict__ ws_kern, const float* __restrict__ syn_e,
    const float* __restrict__ syn_i,  const float* __restrict__ Zin,
    float* __restrict__ subpart, float* __restrict__ spkout) {
  __shared__ float ek[T_NO_], ik[T_NO_], hk[T_NO_], sk[T_NO_];
  __shared__ float we[455], wi[455], wz[455];
  const int s   = blockIdx.y;
  const int t0  = blockIdx.x * 256;
  const int tid = threadIdx.x;
  if (tid < T_NO_) {
    ek[tid] = ws_kern[(0 * SUB_NO_ + s) * T_NO_ + tid];
    ik[tid] = ws_kern[(1 * SUB_NO_ + s) * T_NO_ + tid];
    hk[tid] = ws_kern[(2 * SUB_NO_ + s) * T_NO_ + tid];
    sk[tid] = ws_kern[(3 * SUB_NO_ + s) * T_NO_ + tid];
  }
  for (int i = tid; i < 455; i += 256) {
    int ts = t0 - 199 + i;                       // syn window base
    bool oks = (ts >= 0) && (ts < T_DATA_);
    we[i] = oks ? syn_e[(size_t)ts * SPAD_ + s] : 0.0f;
    wi[i] = oks ? syn_i[(size_t)ts * SPAD_ + s] : 0.0f;
    int tz = t0 - 200 + i;                       // Z window base (1-step delay)
    bool okz = (tz >= 0) && (tz < T_DATA_);
    wz[i] = okz ? Zin[(size_t)tz * SUB_NO_ + s] : 0.0f;
  }
  __syncthreads();
  const int t = t0 + tid;
  if (t >= T_DATA_) return;
  float fe = 0.f, fi = 0.f, fh = 0.f, fs = 0.f;
#pragma unroll 8
  for (int k = 0; k < T_NO_; ++k) {
    const int w = tid + 199 - k;  // syn: t-k ; Z: t-1-k (different window bases)
    const float z = wz[w];
    fe += we[w] * ek[k];
    fi += wi[w] * ik[k];
    fh += z * hk[k];
    fs += z * sk[k];
  }
  subpart[(size_t)t * SPAD_ + s] = fe + fi + fh;
  spkout [(size_t)t * SPAD_ + s] = fs;
}

// ---------------------------------------------------------------------------
// Kernel 4: C_den gather + Theta + leaky-ReLU MLP + sigmoid; final_V.
// ---------------------------------------------------------------------------
__global__ void __launch_bounds__(256) mlp_kernel(
    const float* __restrict__ subpart, const float* __restrict__ spk,
    const float* __restrict__ Cden, const float* __restrict__ Theta,
    const float* __restrict__ Vo,   const float* __restrict__ w1,
    const float* __restrict__ w2,   const float* __restrict__ w3,
    float* __restrict__ outV, float* __restrict__ outZ) {
  __shared__ float sC[SUB_NO_ * SUB_NO_];
  __shared__ float sTh[SUB_NO_], sW1[SUB_NO_ * 6], sW2[36], sW3[6 * SUB_NO_];
  const int tid = threadIdx.x;
  if (tid < SUB_NO_ * SUB_NO_) sC[tid]  = Cden[tid];
  if (tid < SUB_NO_)           sTh[tid] = Theta[tid];
  if (tid < SUB_NO_ * 6)       sW1[tid] = w1[tid];
  if (tid < 36)                sW2[tid] = w2[tid];
  if (tid < 6 * SUB_NO_)       sW3[tid] = w3[tid];
  __syncthreads();
  const int t = blockIdx.x * 256 + tid;
  if (t >= T_DATA_) return;
  float sp[SUB_NO_], sb[SUB_NO_];
#pragma unroll
  for (int j = 0; j < SUB_NO_; ++j) {
    sp[j] = spk    [(size_t)t * SPAD_ + j];
    sb[j] = subpart[(size_t)t * SPAD_ + j];
  }
  outV[t] = sp[0] + Vo[0];
#pragma unroll
  for (int s = 0; s < SUB_NO_; ++s) {
    float x = sb[s] + sTh[s];
#pragma unroll
    for (int j = 0; j < SUB_NO_; ++j) x += sC[s * SUB_NO_ + j] * sp[j];
    float h0 = (x > 0.f) ? x : 0.01f * x;
    float h1[6];
#pragma unroll
    for (int k = 0; k < 6; ++k) {
      float y = h0 * sW1[s * 6 + k];
      h1[k] = (y > 0.f) ? y : 0.01f * y;
    }
    float zacc = 0.f;
#pragma unroll
    for (int j = 0; j < 6; ++j) {
      float acc = 0.f;
#pragma unroll
      for (int k = 0; k < 6; ++k) acc += h1[k] * sW2[k * 6 + j];
      float h2 = (acc > 0.f) ? acc : 0.01f * acc;
      zacc += h2 * sW3[j * SUB_NO_ + s];
    }
    outZ[(size_t)t * SUB_NO_ + s] = 1.0f / (1.0f + __expf(-zacc));
  }
}

// ---------------------------------------------------------------------------
// Host-side launch
// ---------------------------------------------------------------------------
extern "C" void kernel_launch(void* const* d_in, const int* in_sizes, int n_in,
                              void* d_out, int out_size, void* d_ws, size_t ws_size,
                              hipStream_t stream) {
  (void)in_sizes; (void)n_in; (void)out_size; (void)ws_size;
  const float* S_e   = (const float*)d_in[0];
  const float* S_i   = (const float*)d_in[1];
  const float* Z     = (const float*)d_in[2];
  const float* C_den = (const float*)d_in[3];
  const float* C_e   = (const float*)d_in[4];
  const float* C_i   = (const float*)d_in[5];
  const float* Wsyn  = (const float*)d_in[6];
  const float* Whist = (const float*)d_in[7];
  const float* Wspk  = (const float*)d_in[8];
  const float* Delta = (const float*)d_in[9];
  const float* Tau   = (const float*)d_in[10];
  const float* Vo    = (const float*)d_in[11];
  const float* Theta = (const float*)d_in[12];
  const float* w1    = (const float*)d_in[13];
  const float* w2    = (const float*)d_in[14];
  const float* w3    = (const float*)d_in[15];

  float* outV = (float*)d_out;                         // [T]
  float* outZ = outV + T_DATA_;                        // [T, 20]
  float* outF = outZ + (size_t)T_DATA_ * SUB_NO_;      // [80, 200]

  const int nchunks_e = (E_NO_ + 31) / 32;             // 63
  const int nchunks_i = (I_NO_ + 31) / 32;             // 13

  float* ws      = (float*)d_ws;
  float* ws_kern = ws;                                 // 16000 f32
  float* ws_syne = ws_kern + 4 * SUB_NO_ * T_NO_;      // [T, 32]
  float* ws_syni = ws_syne + (size_t)T_DATA_ * SPAD_;  // [T, 32]
  float* ws_subp = ws_syni + (size_t)T_DATA_ * SPAD_;  // [T, 32]
  float* ws_spk  = ws_subp + (size_t)T_DATA_ * SPAD_;  // [T, 32]
  unsigned* ws_bfrag_e = (unsigned*)(ws_spk + (size_t)T_DATA_ * SPAD_);
  unsigned* ws_bfrag_i = ws_bfrag_e + (size_t)2 * nchunks_e * 32 * 8;

  filters_kernel<<<(4 * SUB_NO_ * T_NO_ + 255) / 256, 256, 0, stream>>>(
      Wsyn, Whist, Wspk, Delta, Tau, outF, ws_kern);

  pack_b_kernel<<<(2 * nchunks_e * 32 * 8 + 255) / 256, 256, 0, stream>>>(
      C_e, ws_bfrag_e, E_NO_, nchunks_e);
  pack_b_kernel<<<(2 * nchunks_i * 32 * 8 + 255) / 256, 256, 0, stream>>>(
      C_i, ws_bfrag_i, I_NO_, nchunks_i);

  const int tstrips = T_DATA_ / 32;  // 3125 exact
  agg_wmma_kernel<<<tstrips * 2, 32, 0, stream>>>(S_e, ws_bfrag_e, ws_syne,
                                                  E_NO_, nchunks_e);
  agg_wmma_kernel<<<tstrips * 2, 32, 0, stream>>>(S_i, ws_bfrag_i, ws_syni,
                                                  I_NO_, nchunks_i);

  dim3 cgrid((T_DATA_ + 255) / 256, SUB_NO_);
  conv_kernel<<<cgrid, 256, 0, stream>>>(ws_kern, ws_syne, ws_syni, Z,
                                         ws_subp, ws_spk);

  mlp_kernel<<<(T_DATA_ + 255) / 256, 256, 0, stream>>>(
      ws_subp, ws_spk, C_den, Theta, Vo, w1, w2, w3, outV, outZ);
}
